// lstm_79525614453037
// MI455X (gfx1250) — matrix operand order repaired and checked by reading, not compiled, not run on hardware
//
#include <hip/hip_runtime.h>

typedef __attribute__((ext_vector_type(16))) _Float16 v16h;
typedef __attribute__((ext_vector_type(8)))  _Float16 v8h;
typedef __attribute__((ext_vector_type(8)))  float    v8f;
typedef __attribute__((ext_vector_type(4)))  float    v4f;

#define T_STEPS 256
#define ISZ 8
#define HSZ 32
#define WAVES_PER_BLOCK 4
#define ROWS_PER_WAVE 16            // batch rows per wave (one WMMA M-tile)
#define LDS_ROW 40                  // f16 elems per staged row (32 used + pad), 80B: 16B aligned, bank-spread

// ---- fast nonlinearity: native v_tanh_f32 on gfx1250 if the builtin exists ----
#if __has_builtin(__builtin_amdgcn_tanhf)
__device__ __forceinline__ float tanh_f(float x) {
    return __builtin_amdgcn_tanhf(x);
}
#else
__device__ __forceinline__ float tanh_f(float x) {
    // 1 - 2/(exp(2x)+1), with v_rcp_f32 instead of IEEE divide
    return 1.0f - 2.0f * __builtin_amdgcn_rcpf(__expf(2.0f * x) + 1.0f);
}
#endif
__device__ __forceinline__ float sig_f(float x) {
    return 0.5f * tanh_f(0.5f * x) + 0.5f;
}

#if __has_builtin(__builtin_amdgcn_sched_barrier)
#define SCHED_FENCE() __builtin_amdgcn_sched_barrier(0)
#else
#define SCHED_FENCE()
#endif

__global__ __launch_bounds__(WAVES_PER_BLOCK * 32)
void lstm_wmma_kernel(const float* __restrict__ x,
                      const float* __restrict__ W_ih,
                      const float* __restrict__ W_hh,
                      const float* __restrict__ b_ih,
                      const float* __restrict__ b_hh,
                      const float* __restrict__ W_out,
                      const float* __restrict__ b_out,
                      float* __restrict__ out,
                      int Btotal)
{
    __shared__ _Float16 lds_h[WAVES_PER_BLOCK * ROWS_PER_WAVE * LDS_ROW];

    const int tid  = threadIdx.x;
    const int wave = tid >> 5;
    const int lane = tid & 31;
    const int n16  = lane & 15;   // column-in-tile / A-row index
    const int half = lane >> 4;   // 0 or 1

    const int b0 = (blockIdx.x * WAVES_PER_BLOCK + wave) * ROWS_PER_WAVE;
    if (b0 >= Btotal) return;

    _Float16* hstage = lds_h + wave * ROWS_PER_WAVE * LDS_ROW;

    // ---------------- Preload weights into WMMA B-layout (f16, registers) ----------------
    // B-layout (32x16 f16): lane holds column n = g*16 + n16; element e <-> K = half*16 + e.
    // bx additionally carries the bias (b_ih+b_hh) in its K=8 row (half==0, element 8);
// ax carries a constant 1.0 at K=8 (half==0, element 8), so bias rides the x-WMMA.
    v16h bh[8];           // W_hh^T tiles, K = hidden 0..31
    v16h bx[8];           // W_ih^T tiles zero-padded to K=32 (K<8 = weights, K=8 = bias)
#pragma unroll
    for (int g = 0; g < 8; ++g) {
        const int n = g * 16 + n16;                 // gate index 0..127
        const float* wrow = W_hh + n * HSZ + half * 16;
        v4f w0 = *(const v4f*)(wrow + 0);
        v4f w1 = *(const v4f*)(wrow + 4);
        v4f w2 = *(const v4f*)(wrow + 8);
        v4f w3 = *(const v4f*)(wrow + 12);
#pragma unroll
        for (int e = 0; e < 4; ++e) {
            bh[g][e]      = (_Float16)w0[e];
            bh[g][4 + e]  = (_Float16)w1[e];
            bh[g][8 + e]  = (_Float16)w2[e];
            bh[g][12 + e] = (_Float16)w3[e];
        }
        v16h zv = {};
        bx[g] = zv;
        if (half == 0) {
            const float* wi = W_ih + n * ISZ;       // K = 0..7
            v4f a0 = *(const v4f*)(wi + 0);
            v4f a1 = *(const v4f*)(wi + 4);
#pragma unroll
            for (int e = 0; e < 4; ++e) {
                bx[g][e]     = (_Float16)a0[e];
                bx[g][4 + e] = (_Float16)a1[e];
            }
            bx[g][8] = (_Float16)(b_ih[n] + b_hh[n]);   // bias at K=8
        }
    }

    // ---------------- State ----------------
    v8f c_acc[2] = {};             // c in C/D layout: tile j covers hidden cols j*16..j*16+15
    v16h ah = {};                  // h in A-layout (f16), h0 = 0

    // Persistent A_x: element 8 = 1.0 on half==0 lanes (bias row K=8), other elements >= 8 stay 0;
    // never clobbered because only elements 0..7 of half==0 lanes are written.
    v16h ax = {};
    if (half == 0) ax[8] = (_Float16)1.0f;

    // Per-lane x row base: A-layout row m = n16
    const float* xrow = x + (size_t)(b0 + n16) * T_STEPS * ISZ;
    const v8f zc = {};

    // ---------------- Recurrence over T ----------------
    for (int t = 0; t < T_STEPS; ++t) {
        // A_x: K=0..7 = x_t (half==0 lanes, elements 0..7), updated in place.
        if (half == 0) {
            const float* xp = xrow + t * ISZ;
            v4f p0 = *(const v4f*)xp;
            v4f p1 = *(const v4f*)(xp + 4);
#pragma unroll
            for (int e = 0; e < 4; ++e) {
                ax[e]     = (_Float16)p0[e];
                ax[4 + e] = (_Float16)p1[e];
            }
        }
        if (t + 8 < T_STEPS)
            __builtin_prefetch(xrow + (t + 8) * ISZ, 0, 0);   // global_prefetch_b8

        // Gate pre-activations z[g] for all 8 gate tiles.
        // Phase 1: 8 independent x-projection WMMAs (C = inline 0, bias folded in).
        // Phase 2: 8 h-projection WMMAs (D->C accumulation, no RAW-on-A/B hazard).
        // sched_barrier keeps the phases apart so every D-register read (activations)
        // lands >= 4 co-exec slots after its producing WMMA -> no v_nop hazard fill.
        v8f z[8];
#pragma unroll
        for (int g = 0; g < 8; ++g)
            z[g] = __builtin_amdgcn_wmma_f32_16x16x32_f16(false, ax, false, bx[g], (short)0, zc, false, false);
        SCHED_FENCE();
#pragma unroll
        for (int g = 0; g < 8; ++g)
            z[g] = __builtin_amdgcn_wmma_f32_16x16x32_f16(false, ah, false, bh[g], (short)0, z[g], false, false);

        // Elementwise LSTM cell update in C/D layout; stage new h to LDS (f16).
        // Gate tiles for hidden-column half j: i -> z[0+j], f -> z[2+j], g -> z[4+j], o -> z[6+j]
#pragma unroll
        for (int j = 0; j < 2; ++j) {
#pragma unroll
            for (int r = 0; r < 8; ++r) {
                float iv = sig_f(z[0 + j][r]);
                float fv = sig_f(z[2 + j][r]);
                float gv = tanh_f(z[4 + j][r]);
                float ov = sig_f(z[6 + j][r]);
                float cn = fv * c_acc[j][r] + iv * gv;
                c_acc[j][r] = cn;
                float hv = ov * tanh_f(cn);
                // C/D layout: element r of lane -> (m = r + 8*half, n = j*16 + n16)
                hstage[(r + 8 * half) * LDS_ROW + j * 16 + n16] = (_Float16)hv;
            }
        }

        // Relayout h: C/D-layout (LDS) -> A-layout (registers) for next step.
        // LDS ops are in-order within a wave; wait + compiler barrier for safety.
        asm volatile("s_wait_dscnt 0x0" ::: "memory");
        {
            const _Float16* hb = hstage + n16 * LDS_ROW;     // A row m = n16
            // elements 0..7  <-> K = half*16 + 0..7
            // elements 8..15 <-> K = half*16 + 8 + 0..7
            v8h lo = *(const v8h*)(hb + half * 16);
            v8h hi = *(const v8h*)(hb + half * 16 + 8);
#pragma unroll
            for (int e = 0; e < 8; ++e) {
                ah[e]     = lo[e];
                ah[8 + e] = hi[e];
            }
        }
        asm volatile("" ::: "memory");
    }

    // ---------------- Output projection: out[b] = h_T[b,:] . W_out + b_out ----------------
    asm volatile("s_wait_dscnt 0x0" ::: "memory");
    if (lane < 16) {
        const _Float16* hb = hstage + lane * LDS_ROW;
        float sum = 0.0f;
#pragma unroll
        for (int n = 0; n < HSZ; ++n)
            sum += (float)hb[n] * W_out[n];
        out[b0 + lane] = sum + b_out[0];
    }
}

extern "C" void kernel_launch(void* const* d_in, const int* in_sizes, int n_in,
                              void* d_out, int out_size, void* d_ws, size_t ws_size,
                              hipStream_t stream) {
    const float* x     = (const float*)d_in[0];
    const float* W_ih  = (const float*)d_in[1];
    const float* W_hh  = (const float*)d_in[2];
    const float* b_ih  = (const float*)d_in[3];
    const float* b_hh  = (const float*)d_in[4];
    const float* W_out = (const float*)d_in[5];
    const float* b_out = (const float*)d_in[6];
    float* out = (float*)d_out;

    const int Btotal = in_sizes[0] / (T_STEPS * ISZ);            // 4096
    const int rows_per_block = WAVES_PER_BLOCK * ROWS_PER_WAVE;  // 64
    const int grid = (Btotal + rows_per_block - 1) / rows_per_block;

    lstm_wmma_kernel<<<dim3(grid), dim3(WAVES_PER_BLOCK * 32), 0, stream>>>(
        x, W_ih, W_hh, b_ih, b_hh, W_out, b_out, out, Btotal);
}
